// AttentionModule_27573690040528
// MI455X (gfx1250) — matrix operand order repaired and checked by reading, not compiled, not run on hardware
//
#include <hip/hip_runtime.h>

// ---------------- problem constants ----------------
constexpr int B_ = 2, S_ = 2048, M_ = 2048, E_ = 1024, H_ = 16, D_ = 64;
constexpr float LN_EPS = 1e-5f;

// ---------------- WMMA types (CDNA5 / gfx1250, wave32) ----------------
typedef __bf16 bf16_t;
typedef bf16_t v16bf __attribute__((ext_vector_type(16)));
typedef float  v8f   __attribute__((ext_vector_type(8)));

union BF16Frag {
    v16bf v;
    unsigned short u[16];
    uint4 q[2];          // two 16B halves -> global_load_b128 / ds_load_b128
};

__device__ __forceinline__ unsigned short f2bf(float f) {
    union { float f; unsigned int u; } x; x.f = f;
    unsigned int u = x.u;
    return (unsigned short)((u + 0x7FFFu + ((u >> 16) & 1u)) >> 16);  // RNE
}

__device__ __forceinline__ v8f wmma_bf16(const BF16Frag& a, const BF16Frag& b, v8f c) {
    // D = A(16x32 bf16) x B(32x16 bf16) + C(16x16 f32)
    return __builtin_amdgcn_wmma_f32_16x16x32_bf16(
        /*neg_a=*/false, a.v, /*neg_b=*/false, b.v,
        /*c_mod=*/(short)0, c, /*reuse_a=*/false, /*reuse_b=*/false);
}

// ============================================================
// Kernel 1: projection GEMM  O = (X @ W^T + bias) * scale   (bf16 out)
// X:[rows,E] f32, W:[E,E] f32 row-major ([f][e]); rows = 4096.
// WG: 256 threads (8 waves), tile 128(M) x 64(N), K-step 32.
// vt==0: store O[row*E+col].  vt==1: store head-transposed Vt[b,h,d,m].
// ============================================================
__global__ __launch_bounds__(256) void proj_gemm(
    const float* __restrict__ X, const float* __restrict__ W,
    const float* __restrict__ bias, unsigned short* __restrict__ Obf,
    float scale, int vt)
{
    __shared__ unsigned short lX[128 * 32];  // [m][k] bf16, 8 KB
    __shared__ unsigned short lW[64 * 32];   // [n][k] bf16, 4 KB

    const int tid  = threadIdx.x;
    const int wave = tid >> 5, lane = tid & 31;
    const int lh = lane >> 4, ln = lane & 15;
    const int row0 = blockIdx.x * 128;
    const int n0   = blockIdx.y * 64;

    v8f acc[4] = {};

    for (int k0 = 0; k0 < E_; k0 += 32) {
        __syncthreads();
        for (int idx = tid; idx < 128 * 32; idx += 256) {
            int r = idx >> 5, kk = idx & 31;
            lX[idx] = f2bf(X[(size_t)(row0 + r) * E_ + k0 + kk]);
        }
        for (int idx = tid; idx < 64 * 32; idx += 256) {
            int n = idx >> 5, kk = idx & 31;
            lW[idx] = f2bf(W[(size_t)(n0 + n) * E_ + k0 + kk]);
        }
        if (k0 + 32 < E_)  // next-tile hint (global_prefetch_b8)
            __builtin_prefetch(&X[(size_t)(row0 + (tid >> 1)) * E_ + k0 + 32], 0, 0);
        __syncthreads();

        // A fragment: this wave's 16 rows; both 8-element runs are contiguous
        BF16Frag a;
        const unsigned short* ap = &lX[(wave * 16 + ln) * 32];
        a.q[0] = *(const uint4*)(ap + 8 * lh);
        a.q[1] = *(const uint4*)(ap + 16 + 8 * lh);

#pragma unroll
        for (int t = 0; t < 4; t++) {
            BF16Frag b;
            const unsigned short* bp = &lW[(t * 16 + ln) * 32];
            b.q[0] = *(const uint4*)(bp + 8 * lh);
            b.q[1] = *(const uint4*)(bp + 16 + 8 * lh);
            acc[t] = wmma_bf16(a, b, acc[t]);
        }
    }

    // epilogue: bias, scale, convert to bf16
#pragma unroll
    for (int t = 0; t < 4; t++) {
        const int col = n0 + t * 16 + ln;
        const float bv = bias[col];
#pragma unroll
        for (int r = 0; r < 8; r++) {
            const int row = row0 + wave * 16 + r + 8 * lh;
            const unsigned short o = f2bf((acc[t][r] + bv) * scale);
            if (vt == 0) {
                Obf[(size_t)row * E_ + col] = o;
            } else {
                // Vt[((b*H + h)*D + d)*M + m]  (b=row/S, m=row%S, h=col/D, d=col%D)
                const int bb = row >> 11, m = row & (S_ - 1);
                const int hh = col >> 6,  dd = col & (D_ - 1);
                Obf[(((size_t)(bb * H_ + hh)) * D_ + dd) * M_ + m] = o;
            }
        }
    }
}

// ============================================================
// Kernel 2: fused attention for one (batch, 16-row query tile).
// Loops over heads; scores & running head-avg live in LDS (2x128KB,
// only possible with CDNA5's 320KB/WGP LDS).
// ============================================================
__global__ __launch_bounds__(256) void attn_kernel(
    const unsigned short* __restrict__ Qbf,   // [B,S,E] bf16 (pre-scaled 1/sqrt(D))
    const unsigned short* __restrict__ Kbf,   // [B,M,E] bf16
    const unsigned short* __restrict__ Vt,    // [B,H,D,M] bf16 (head-transposed)
    unsigned short* __restrict__ Abf,         // attended out [B,S,E] bf16
    float* __restrict__ avg_out)              // [B,S,M] f32
{
    extern __shared__ char smem[];
    float* sS   = (float*)smem;          // [16][M] scores -> probs (128 KB)
    float* sAvg = sS + 16 * M_;          // [16][M] head-summed probs (128 KB)
    float* sRed = sAvg + 16 * M_;        // [8][16][16] PV partials (8 KB)

    const int tid = threadIdx.x;
    const int wave = tid >> 5, lane = tid & 31;
    const int lh = lane >> 4, ln = lane & 15;
    const int b  = blockIdx.x / (S_ / 16);
    const int s0 = (blockIdx.x % (S_ / 16)) * 16;

    for (int i = tid; i < 16 * M_; i += 256) sAvg[i] = 0.f;
    __syncthreads();

    for (int h = 0; h < H_; h++) {
        // ---- scores(16 x M) = Qtile(16 x 64) x K^T(64 x M) ----
        BF16Frag a0, a1;
        const unsigned short* qrow = Qbf + ((size_t)(b * S_ + s0 + ln)) * E_ + h * D_;
        a0.q[0] = *(const uint4*)(qrow + 8 * lh);
        a0.q[1] = *(const uint4*)(qrow + 16 + 8 * lh);
        a1.q[0] = *(const uint4*)(qrow + 32 + 8 * lh);
        a1.q[1] = *(const uint4*)(qrow + 48 + 8 * lh);

        for (int nt = wave; nt < M_ / 16; nt += 8) {
            const int m0 = nt * 16;
            v8f c = {};
            // B[k][n] = K[m0+n][h*D + k]; lane: n=ln, k = 8*lh+e / 16+8*lh+(e-8)  -> table order
            const unsigned short* krow =
                Kbf + ((size_t)(b * M_ + m0 + ln)) * E_ + h * D_;
            BF16Frag b0, b1;
            b0.q[0] = *(const uint4*)(krow + 8 * lh);
            b0.q[1] = *(const uint4*)(krow + 16 + 8 * lh);
            b1.q[0] = *(const uint4*)(krow + 32 + 8 * lh);
            b1.q[1] = *(const uint4*)(krow + 48 + 8 * lh);
            c = wmma_bf16(a0, b0, c);
            c = wmma_bf16(a1, b1, c);
#pragma unroll
            for (int r = 0; r < 8; r++) sS[(r + 8 * lh) * M_ + m0 + ln] = c[r];
        }
        __syncthreads();

        // ---- softmax per row (wave handles 2 rows; wave32 shuffles) ----
        for (int rr = 0; rr < 2; rr++) {
            const int row = wave * 2 + rr;
            float mx = -3.402823466e38f;
            for (int j = lane; j < M_; j += 32) mx = fmaxf(mx, sS[row * M_ + j]);
            for (int o = 16; o > 0; o >>= 1) mx = fmaxf(mx, __shfl_xor(mx, o, 32));
            float sum = 0.f;
            for (int j = lane; j < M_; j += 32) {
                float e = __expf(sS[row * M_ + j] - mx);
                sS[row * M_ + j] = e;
                sum += e;
            }
            for (int o = 16; o > 0; o >>= 1) sum += __shfl_xor(sum, o, 32);
            const float inv = 1.f / sum;
            for (int j = lane; j < M_; j += 32) {
                float p = sS[row * M_ + j] * inv;
                sS[row * M_ + j] = p;
                sAvg[row * M_ + j] += p;
            }
        }
        __syncthreads();

        // ---- attended(16 x 64) = P(16 x M) x V(M x 64) ----
        // waves 0-3 / 4-7 split the K range; n-tile (d block) = wave & 3
        const int ntv = wave & 3;
        const int kbeg = (wave >> 2) * (M_ / 2);
        v8f c = {};
        for (int k0 = kbeg; k0 < kbeg + M_ / 2; k0 += 32) {
            // A: probs row ln, k runs contiguous -> float4 LDS loads + cvt
            BF16Frag a;
            const float* pp = &sS[ln * M_ + k0];
            const float4 f0 = *(const float4*)(pp + 8 * lh);
            const float4 f1 = *(const float4*)(pp + 8 * lh + 4);
            const float4 f2 = *(const float4*)(pp + 16 + 8 * lh);
            const float4 f3 = *(const float4*)(pp + 16 + 8 * lh + 4);
            a.u[0] = f2bf(f0.x); a.u[1] = f2bf(f0.y); a.u[2] = f2bf(f0.z); a.u[3] = f2bf(f0.w);
            a.u[4] = f2bf(f1.x); a.u[5] = f2bf(f1.y); a.u[6] = f2bf(f1.z); a.u[7] = f2bf(f1.w);
            a.u[8]  = f2bf(f2.x); a.u[9]  = f2bf(f2.y); a.u[10] = f2bf(f2.z); a.u[11] = f2bf(f2.w);
            a.u[12] = f2bf(f3.x); a.u[13] = f2bf(f3.y); a.u[14] = f2bf(f3.z); a.u[15] = f2bf(f3.w);
            // B[k][n] = Vt[b,h, d = ntv*16+ln, m = k0 + table-order k] -> two b128 runs
            BF16Frag bb;
            const unsigned short* vrow =
                Vt + (((size_t)(b * H_ + h)) * D_ + ntv * 16 + ln) * M_ + k0;
            bb.q[0] = *(const uint4*)(vrow + 8 * lh);
            bb.q[1] = *(const uint4*)(vrow + 16 + 8 * lh);
            c = wmma_bf16(a, bb, c);
        }
#pragma unroll
        for (int r = 0; r < 8; r++) sRed[wave * 256 + (r + 8 * lh) * 16 + ln] = c[r];
        __syncthreads();
        if (wave < 4) {
#pragma unroll
            for (int r = 0; r < 8; r++) {
                const int row = r + 8 * lh;
                const float v = sRed[wave * 256 + row * 16 + ln] +
                                sRed[(wave + 4) * 256 + row * 16 + ln];
                Abf[((size_t)(b * S_ + s0 + row)) * E_ + h * D_ + wave * 16 + ln] = f2bf(v);
            }
        }
        __syncthreads();
    }

    // ---- write avg_attention = sum_h P / H ----
    const float invH = 1.f / (float)H_;
    for (int i = tid; i < 16 * M_; i += 256) {
        const int row = i / M_, col = i % M_;
        avg_out[((size_t)(b * S_ + s0 + row)) * M_ + col] = sAvg[i] * invH;
    }
}

// ============================================================
// Kernel 3: output projection + bias + residual -> f32 Y
// ============================================================
__global__ __launch_bounds__(256) void oproj_gemm(
    const unsigned short* __restrict__ Abf,  // [rows,E] bf16
    const float* __restrict__ Wo, const float* __restrict__ bo,
    const float* __restrict__ resid, float* __restrict__ Y)
{
    __shared__ unsigned short lX[128 * 32];
    __shared__ unsigned short lW[64 * 32];

    const int tid = threadIdx.x;
    const int wave = tid >> 5, lane = tid & 31;
    const int lh = lane >> 4, ln = lane & 15;
    const int row0 = blockIdx.x * 128;
    const int n0   = blockIdx.y * 64;

    v8f acc[4] = {};

    for (int k0 = 0; k0 < E_; k0 += 32) {
        __syncthreads();
        for (int idx = tid; idx < 128 * 32; idx += 256) {
            int r = idx >> 5, kk = idx & 31;
            lX[idx] = Abf[(size_t)(row0 + r) * E_ + k0 + kk];
        }
        for (int idx = tid; idx < 64 * 32; idx += 256) {
            int n = idx >> 5, kk = idx & 31;
            lW[idx] = f2bf(Wo[(size_t)(n0 + n) * E_ + k0 + kk]);
        }
        __syncthreads();

        BF16Frag a;
        const unsigned short* ap = &lX[(wave * 16 + ln) * 32];
        a.q[0] = *(const uint4*)(ap + 8 * lh);
        a.q[1] = *(const uint4*)(ap + 16 + 8 * lh);
#pragma unroll
        for (int t = 0; t < 4; t++) {
            BF16Frag b;
            const unsigned short* bp = &lW[(t * 16 + ln) * 32];
            b.q[0] = *(const uint4*)(bp + 8 * lh);
            b.q[1] = *(const uint4*)(bp + 16 + 8 * lh);
            acc[t] = wmma_bf16(a, b, acc[t]);
        }
    }

#pragma unroll
    for (int t = 0; t < 4; t++) {
        const int col = n0 + t * 16 + ln;
        const float bv = bo[col];
#pragma unroll
        for (int r = 0; r < 8; r++) {
            const int row = row0 + wave * 16 + r + 8 * lh;
            Y[(size_t)row * E_ + col] = acc[t][r] + bv + resid[(size_t)row * E_ + col];
        }
    }
}

// ============================================================
// Kernel 4: LayerNorm over E per row
// ============================================================
__global__ __launch_bounds__(256) void ln_kernel(
    const float* __restrict__ Y, const float* __restrict__ gamma,
    const float* __restrict__ beta, float* __restrict__ out)
{
    const int row = blockIdx.x;
    const float* y = Y + (size_t)row * E_;
    __shared__ float r1[8], r2[8];
    const int tid = threadIdx.x, wave = tid >> 5, lane = tid & 31;

    float s1 = 0.f, s2 = 0.f;
    for (int j = tid; j < E_; j += 256) { float v = y[j]; s1 += v; s2 += v * v; }
    for (int o = 16; o > 0; o >>= 1) { s1 += __shfl_xor(s1, o, 32); s2 += __shfl_xor(s2, o, 32); }
    if (lane == 0) { r1[wave] = s1; r2[wave] = s2; }
    __syncthreads();
    float t1 = 0.f, t2 = 0.f;
#pragma unroll
    for (int w = 0; w < 8; w++) { t1 += r1[w]; t2 += r2[w]; }
    const float mu  = t1 * (1.f / E_);
    const float var = t2 * (1.f / E_) - mu * mu;
    const float inv = rsqrtf(var + LN_EPS);
    float* o = out + (size_t)row * E_;
    for (int j = tid; j < E_; j += 256)
        o[j] = (y[j] - mu) * inv * gamma[j] + beta[j];
}

// ============================================================
extern "C" void kernel_launch(void* const* d_in, const int* in_sizes, int n_in,
                              void* d_out, int out_size, void* d_ws, size_t ws_size,
                              hipStream_t stream)
{
    (void)in_sizes; (void)n_in; (void)out_size; (void)ws_size;

    const float* query  = (const float*)d_in[0];
    const float* keys   = (const float*)d_in[1];
    const float* values = (const float*)d_in[2];
    const float* Wq = (const float*)d_in[3];
    const float* bq = (const float*)d_in[4];
    const float* Wk = (const float*)d_in[5];
    const float* bk = (const float*)d_in[6];
    const float* Wv = (const float*)d_in[7];
    const float* bv = (const float*)d_in[8];
    const float* Wo = (const float*)d_in[9];
    const float* bo = (const float*)d_in[10];
    const float* gamma = (const float*)d_in[11];
    const float* beta  = (const float*)d_in[12];

    const size_t nTok = (size_t)B_ * S_;  // 4096 (== B*M)
    char* ws = (char*)d_ws;
    unsigned short* Qbf = (unsigned short*)ws; ws += nTok * E_ * sizeof(unsigned short);
    unsigned short* Kbf = (unsigned short*)ws; ws += nTok * E_ * sizeof(unsigned short);
    unsigned short* Vt  = (unsigned short*)ws; ws += nTok * E_ * sizeof(unsigned short);
    unsigned short* Abf = (unsigned short*)ws; ws += nTok * E_ * sizeof(unsigned short);
    float* Yf = (float*)ws;  // [B,S,E] f32

    float* out_ln  = (float*)d_out;            // [B,S,E]
    float* out_avg = out_ln + nTok * E_;       // [B,S,M]

    dim3 gproj(nTok / 128, E_ / 64);
    // Q pre-scaled by 1/sqrt(D) = 0.125
    proj_gemm<<<gproj, 256, 0, stream>>>(query,  Wq, bq, Qbf, 0.125f, 0);
    proj_gemm<<<gproj, 256, 0, stream>>>(keys,   Wk, bk, Kbf, 1.0f, 0);
    proj_gemm<<<gproj, 256, 0, stream>>>(values, Wv, bv, Vt,  1.0f, 1);  // head-transposed

    const size_t attn_smem = (size_t)16 * M_ * sizeof(float) * 2   // scores + avg
                           + (size_t)8 * 256 * sizeof(float);      // PV reduction
    attn_kernel<<<B_ * (S_ / 16), 256, attn_smem, stream>>>(Qbf, Kbf, Vt, Abf, out_avg);

    oproj_gemm<<<gproj, 256, 0, stream>>>(Abf, Wo, bo, query, Yf);

    ln_kernel<<<nTok, 256, 0, stream>>>(Yf, gamma, beta, out_ln);
}
